// ChebyNet_37873021616189
// MI455X (gfx1250) — matrix operand order repaired and checked by reading, compiled, not run on hardware
//
#include <hip/hip_runtime.h>

// ---------------------------------------------------------------------------
// ChebyNet (2x ChebConv K=4) for MI455X / gfx1250.
// Sparse propagation: 1 thread per (edge,feature) -> wave-uniform edge data,
//   coalesced gathers, global_atomic_add_f32 scatters (L2-resident: Tx
//   tensors are 6.4-12.8 MB vs 192 MB L2).
// Dense combine: v_wmma_f32_16x16x32_f16, one wave per 16-node tile.
//   Layer 1: 4 WMMAs (one per Chebyshev order, K=F_IN=32 each) into one C.
//   Layer 2: 2 WMMAs (two orders packed per K=32; N padded 10->16).
// ---------------------------------------------------------------------------

typedef __attribute__((ext_vector_type(16))) _Float16 v16h;
typedef __attribute__((ext_vector_type(8)))  float    v8f;

#define N_NODES 100000
#define N_EDGES 1600000
#define F_IN    32
#define HID     16
#define C_OUT   10

// ---------------- graph normalization ----------------

__global__ void degree_kernel(const long long* __restrict__ dst,
                              float* __restrict__ deg, int E) {
    int e = blockIdx.x * blockDim.x + threadIdx.x;
    if (e < E) atomicAdd(&deg[(int)dst[e]], 1.0f);
}

__global__ void dis_kernel(const float* __restrict__ deg,
                           float* __restrict__ dis, int n) {
    int i = blockIdx.x * blockDim.x + threadIdx.x;
    if (i < n) {
        float d = deg[i];
        dis[i] = (d > 0.0f) ? rsqrtf(fmaxf(d, 1.0f)) : 0.0f;
    }
}

__global__ void norm_kernel(const long long* __restrict__ src,
                            const long long* __restrict__ dst,
                            const float* __restrict__ dis,
                            float* __restrict__ nrm, int E) {
    int e = blockIdx.x * blockDim.x + threadIdx.x;
    if (e < E) nrm[e] = -dis[(int)src[e]] * dis[(int)dst[e]];
}

// out[i] = -a[i]  (seed for Tx_k = 2*prop(Tx_{k-1}) - Tx_{k-2})
__global__ void neg_copy_kernel(const float* __restrict__ a,
                                float* __restrict__ out, int n) {
    int i = blockIdx.x * blockDim.x + threadIdx.x;
    if (i < n) out[i] = -a[i];
}

// out[dst[e]*F + f] += scale * norm[e] * v[src[e]*F + f]
// One thread per (edge, feature): edge index is wave-uniform for F=32,
// gather/scatter are contiguous within the wave; atomics land in L2.
__global__ void prop_kernel(const float* __restrict__ v,
                            const long long* __restrict__ src,
                            const long long* __restrict__ dst,
                            const float* __restrict__ nrm,
                            float* __restrict__ out,
                            int E, int F, float scale) {
    int tid = blockIdx.x * blockDim.x + threadIdx.x;
    int e = tid / F;
    int f = tid - e * F;
    if (e < E) {
        int s = (int)src[e];
        int d = (int)dst[e];
        float w = scale * nrm[e];
        atomicAdd(&out[d * F + f], w * v[s * F + f]);
    }
}

// ---------------- WMMA GEMMs ----------------
// Layout helpers (CDNA5 ISA 7.12.2, wave32):
//   A 16x32 f16 : lane l holds row (l&15); half j -> K = (j<8 ? 8*(l>>4)+j
//                                                        : 16 + 8*(l>>4)+j-8)
//   B 32x16 f16 : lane l holds col (l&15); half j -> K = same order as A (a_kidx)
//   C/D 16x16 f32: VGPR r -> row 8*(l>>4)+r, col (l&15)

__device__ __forceinline__ int a_kidx(int j, int hi) {
    return (j < 8) ? (hi * 8 + j) : (16 + hi * 8 + (j - 8));
}

// H = relu( sum_k Tx_k @ W1[k] + b1 ),  Tx_k: [N,32], W1[k]: [32,16]
__global__ void __launch_bounds__(32)
cheb_gemm1_kernel(const float* __restrict__ Tx0, const float* __restrict__ Tx1,
                  const float* __restrict__ Tx2, const float* __restrict__ Tx3,
                  const float* __restrict__ W,   // [4][32][16]
                  const float* __restrict__ bias, // [16]
                  float* __restrict__ H) {
    const int tile = blockIdx.x;
    const int lane = threadIdx.x;
    const int m  = lane & 15;
    const int hi = lane >> 4;
    const int row = tile * 16 + m;

    const float* tx[4] = { Tx0, Tx1, Tx2, Tx3 };
    v8f c = {};

#pragma unroll
    for (int k = 0; k < 4; ++k) {
        const float* __restrict__ trow = tx[k] + (size_t)row * F_IN;
        const float* __restrict__ wk   = W + k * F_IN * HID;
        v16h a, b;
#pragma unroll
        for (int j = 0; j < 16; ++j) {
            a[j] = (_Float16)trow[a_kidx(j, hi)];
            b[j] = (_Float16)wk[a_kidx(j, hi) * HID + m];
        }
        c = __builtin_amdgcn_wmma_f32_16x16x32_f16(
                false, a, false, b, (short)0, c, false, false);
    }

    const float bn = bias[m];
#pragma unroll
    for (int r = 0; r < 8; ++r) {
        float v = c[r] + bn;
        H[(size_t)(tile * 16 + hi * 8 + r) * HID + m] = (v > 0.0f) ? v : 0.0f;
    }
}

// logits = sum_k U_k @ W2[k]   (U_k: [N,16], W2[k]: [16,10] zero-padded to 16)
// Two Chebyshev orders packed per WMMA: A = [U_{2p} | U_{2p+1}] (16x32),
// B = [W2[2p] ; W2[2p+1]] (32x16).
__global__ void __launch_bounds__(32)
cheb_gemm2_kernel(const float* __restrict__ T0, const float* __restrict__ T1,
                  const float* __restrict__ T2, const float* __restrict__ T3,
                  const float* __restrict__ W,   // [4][16][10]
                  float* __restrict__ logits) {  // [N][16]
    const int tile = blockIdx.x;
    const int lane = threadIdx.x;
    const int m  = lane & 15;
    const int hi = lane >> 4;
    const int row = tile * 16 + m;

    const float* TA[2] = { T0, T2 };
    const float* TB[2] = { T1, T3 };
    v8f c = {};

#pragma unroll
    for (int p = 0; p < 2; ++p) {
        const float* __restrict__ ra = TA[p] + (size_t)row * HID;
        const float* __restrict__ rb = TB[p] + (size_t)row * HID;
        const float* __restrict__ wa = W + (2 * p)     * HID * C_OUT;
        const float* __restrict__ wb = W + (2 * p + 1) * HID * C_OUT;
        v16h a, b;
#pragma unroll
        for (int j = 0; j < 16; ++j) {
            int ka = a_kidx(j, hi);
            a[j] = (_Float16)((ka < 16) ? ra[ka] : rb[ka - 16]);
            int kb = a_kidx(j, hi);
            float bv = 0.0f;
            if (m < C_OUT)
                bv = (kb < 16) ? wa[kb * C_OUT + m] : wb[(kb - 16) * C_OUT + m];
            b[j] = (_Float16)bv;
        }
        c = __builtin_amdgcn_wmma_f32_16x16x32_f16(
                false, a, false, b, (short)0, c, false, false);
    }

#pragma unroll
    for (int r = 0; r < 8; ++r)
        logits[(size_t)(tile * 16 + hi * 8 + r) * 16 + m] = c[r];
}

// out[n,c] = log_softmax(logits[n,0:10] + b2)
__global__ void logsoftmax_kernel(const float* __restrict__ logits,
                                  const float* __restrict__ bias,
                                  float* __restrict__ out, int n) {
    int i = blockIdx.x * blockDim.x + threadIdx.x;
    if (i >= n) return;
    float v[C_OUT];
    float mx = -3.0e38f;
#pragma unroll
    for (int c = 0; c < C_OUT; ++c) {
        v[c] = logits[(size_t)i * 16 + c] + bias[c];
        mx = fmaxf(mx, v[c]);
    }
    float s = 0.0f;
#pragma unroll
    for (int c = 0; c < C_OUT; ++c) s += expf(v[c] - mx);
    float ls = logf(s);
#pragma unroll
    for (int c = 0; c < C_OUT; ++c) out[(size_t)i * C_OUT + c] = v[c] - mx - ls;
}

// ---------------- driver ----------------

extern "C" void kernel_launch(void* const* d_in, const int* in_sizes, int n_in,
                              void* d_out, int out_size, void* d_ws, size_t ws_size,
                              hipStream_t stream) {
    const float*     x   = (const float*)d_in[0];
    const long long* ei  = (const long long*)d_in[1];   // int64 [2][E]
    const float*     W1  = (const float*)d_in[2];       // [4][32][16]
    const float*     b1  = (const float*)d_in[3];       // [16]
    const float*     W2  = (const float*)d_in[4];       // [4][16][10]
    const float*     b2  = (const float*)d_in[5];       // [10]
    const long long* src = ei;
    const long long* dst = ei + N_EDGES;

    // workspace carve-up (floats)
    float* ws  = (float*)d_ws;
    float* deg = ws;                                 // N
    float* dis = deg + N_NODES;                      // N
    float* nrm = dis + N_NODES;                      // E
    float* Tx1 = nrm + N_EDGES;                      // 32N
    float* Tx2 = Tx1 + (size_t)32 * N_NODES;         // 32N
    float* Tx3 = Tx2 + (size_t)32 * N_NODES;         // 32N
    float* H   = Tx3 + (size_t)32 * N_NODES;         // 16N
    // layer-2 buffers reuse the dead Tx1..Tx3 region
    float* U1  = Tx1;                                // 16N
    float* U2  = Tx1 + (size_t)16 * N_NODES;         // 16N
    float* U3  = Tx1 + (size_t)32 * N_NODES;         // 16N
    float* LG  = Tx1 + (size_t)48 * N_NODES;         // 16N (padded logits)

    const int T = 256;
    const int gE  = (N_EDGES + T - 1) / T;
    const int gN  = (N_NODES + T - 1) / T;
    const int g1  = (N_EDGES * F_IN + T - 1) / T;    // 51.2M threads
    const int g2  = (N_EDGES * HID  + T - 1) / T;    // 25.6M threads
    const int gNF = (N_NODES * F_IN + T - 1) / T;
    const int gNH = (N_NODES * HID  + T - 1) / T;
    const int gTile = N_NODES / 16;                   // 6250, exact

    // normalization: deg -> dis -> norm
    hipMemsetAsync(deg, 0, (size_t)N_NODES * sizeof(float), stream);
    degree_kernel<<<gE, T, 0, stream>>>(dst, deg, N_EDGES);
    dis_kernel<<<gN, T, 0, stream>>>(deg, dis, N_NODES);
    norm_kernel<<<gE, T, 0, stream>>>(src, dst, dis, nrm, N_EDGES);

    // ---- layer 1 (F=32) ----
    hipMemsetAsync(Tx1, 0, (size_t)N_NODES * F_IN * sizeof(float), stream);
    prop_kernel<<<g1, T, 0, stream>>>(x,   src, dst, nrm, Tx1, N_EDGES, F_IN, 1.0f);
    neg_copy_kernel<<<gNF, T, 0, stream>>>(x,   Tx2, N_NODES * F_IN);
    prop_kernel<<<g1, T, 0, stream>>>(Tx1, src, dst, nrm, Tx2, N_EDGES, F_IN, 2.0f);
    neg_copy_kernel<<<gNF, T, 0, stream>>>(Tx1, Tx3, N_NODES * F_IN);
    prop_kernel<<<g1, T, 0, stream>>>(Tx2, src, dst, nrm, Tx3, N_EDGES, F_IN, 2.0f);
    cheb_gemm1_kernel<<<gTile, 32, 0, stream>>>(x, Tx1, Tx2, Tx3, W1, b1, H);

    // ---- layer 2 (F=16) ----
    hipMemsetAsync(U1, 0, (size_t)N_NODES * HID * sizeof(float), stream);
    prop_kernel<<<g2, T, 0, stream>>>(H,  src, dst, nrm, U1, N_EDGES, HID, 1.0f);
    neg_copy_kernel<<<gNH, T, 0, stream>>>(H,  U2, N_NODES * HID);
    prop_kernel<<<g2, T, 0, stream>>>(U1, src, dst, nrm, U2, N_EDGES, HID, 2.0f);
    neg_copy_kernel<<<gNH, T, 0, stream>>>(U1, U3, N_NODES * HID);
    prop_kernel<<<g2, T, 0, stream>>>(U2, src, dst, nrm, U3, N_EDGES, HID, 2.0f);
    cheb_gemm2_kernel<<<gTile, 32, 0, stream>>>(H, U1, U2, U3, W2, LG);

    logsoftmax_kernel<<<gN, T, 0, stream>>>(LG, b2, (float*)d_out, N_NODES);
}
